// SKLinear_triton_55113020342560
// MI455X (gfx1250) — matrix operand order repaired and checked by reading, not compiled, not run on hardware
//
#include <hip/hip_runtime.h>
#include <stdint.h>

// ---------------------------------------------------------------------------
// Types for CDNA5 WMMA (wave32): A/B = 16 bf16 per lane, C/D = 8 f32 per lane
// ---------------------------------------------------------------------------
typedef __bf16 v16bf __attribute__((ext_vector_type(16)));
typedef float  v8f   __attribute__((ext_vector_type(8)));
typedef unsigned int u32x4 __attribute__((ext_vector_type(4)));
typedef int i32x4 __attribute__((ext_vector_type(4)));
typedef int i32x8 __attribute__((ext_vector_type(8)));

union FragBF { v16bf v; u32x4 q[2]; };

#if defined(__has_builtin)
#if __has_builtin(__builtin_amdgcn_tensor_load_to_lds)
#define USE_TDM 1
#else
#define USE_TDM 0
#endif
#else
#define USE_TDM 0
#endif

__device__ __forceinline__ unsigned short f2bf(float f) {
  union { float f; unsigned int u; } x; x.f = f;
  unsigned int r = x.u + 0x7FFFu + ((x.u >> 16) & 1u);  // round-to-nearest-even
  return (unsigned short)(r >> 16);
}

__device__ __forceinline__ void wait_tensorcnt0() {
#if defined(__has_builtin)
#if __has_builtin(__builtin_amdgcn_s_wait_tensorcnt)
  __builtin_amdgcn_s_wait_tensorcnt(0);
  return;
#else
  asm volatile("s_wait_tensorcnt 0x0" ::: "memory");
  return;
#endif
#else
  asm volatile("s_wait_tensorcnt 0x0" ::: "memory");
#endif
}

// ---------------------------------------------------------------------------
// 1) fp32 -> bf16 elementwise pack (8 elems / thread)
// ---------------------------------------------------------------------------
__global__ __launch_bounds__(256) void cvt_bf16_kernel(const float* __restrict__ in,
                                                       unsigned short* __restrict__ out) {
  size_t i = ((size_t)blockIdx.x * 256 + threadIdx.x) * 8;
  union { unsigned short s[8]; u32x4 q; } o;
#pragma unroll
  for (int j = 0; j < 8; ++j) o.s[j] = f2bf(in[i + j]);
  *(u32x4*)(out + i) = o.q;
}

// ---------------------------------------------------------------------------
// 2) Tiled transpose + bf16 pack:
//    dst[(dstRowOff + c) * dstStride + dstColOff + r] = bf16(src[r*C + c])
// ---------------------------------------------------------------------------
__global__ __launch_bounds__(256) void transpose_pack_kernel(
    const float* __restrict__ src, unsigned short* __restrict__ dst,
    int C, int dstStride, int dstRowOff, int dstColOff) {
  __shared__ float tile[32][33];
  const int tx = threadIdx.x & 31;
  const int ty = threadIdx.x >> 5;
  const int r0 = blockIdx.x * 32;
  const int c0 = blockIdx.y * 32;
#pragma unroll
  for (int i = 0; i < 4; ++i)
    tile[ty + i * 8][tx] = src[(size_t)(r0 + ty + i * 8) * C + (c0 + tx)];
  __syncthreads();
#pragma unroll
  for (int i = 0; i < 4; ++i) {
    const int c = c0 + ty + i * 8;
    const int r = r0 + tx;
    dst[(size_t)(dstRowOff + c) * dstStride + (dstColOff + r)] = f2bf(tile[tx][ty + i * 8]);
  }
}

// ---------------------------------------------------------------------------
// TDM: one-instruction 2-D tile DMA, global -> LDS, with row padding that
// reproduces the LDSW = BK+8 layout.  D# per CDNA5 ISA ch.8.
// ---------------------------------------------------------------------------
#if USE_TDM
__device__ __forceinline__ void tdm_load_tile_2d(
    unsigned int lds_byte_addr,   // LDS offset of tile start
    const void* gsrc,             // global address of tile start
    unsigned int rows,            // tile_dim1
    unsigned int row_elems,       // tile_dim0 (bf16 elems per row)
    unsigned long long row_stride_elems,  // tensor_dim0_stride (elements)
    unsigned int pad_interval_code,       // pad every 2^(c+1) DWORDs
    unsigned int pad_amount_code)         // pad (c+1) DWORDs
{
  unsigned long long ga = (unsigned long long)(uintptr_t)gsrc;
  u32x4 g0;
  g0[0] = 1u;                                   // count=1 (valid), user mode
  g0[1] = lds_byte_addr;                        // lds_addr
  g0[2] = (unsigned int)ga;                     // global_addr[31:0]
  g0[3] = (unsigned int)(ga >> 32) | (2u << 30);// global_addr[56:32] | type=2
  const unsigned int td0 = 0x40000000u;         // huge tensor dims: tiles never OOB
  const unsigned int td1 = 0x40000000u;
  i32x8 g1;
  g1[0] = (int)((1u << 16)                      // data_size = 2 bytes
              | (1u << 20)                      // pad_enable
              | (pad_interval_code << 22)
              | (pad_amount_code << 25));       // workgroup_mask[15:0] = 0 (no cluster)
  g1[1] = (int)((td0 & 0xFFFFu) << 16);         // atomic_barrier_addr=0 | tensor_dim0 lo
  g1[2] = (int)((td0 >> 16) | ((td1 & 0xFFFFu) << 16));
  g1[3] = (int)((td1 >> 16) | (row_elems << 16));   // tile_dim0
  g1[4] = (int)rows;                            // tile_dim1 | tile_dim2=0
  g1[5] = (int)(unsigned int)row_stride_elems;  // tensor_dim0_stride lo
  g1[6] = (int)(unsigned int)(row_stride_elems >> 32); // stride hi | dim1_stride lo = 0
  g1[7] = 0;
  const i32x4 z4 = {0, 0, 0, 0};
#if __clang_major__ >= 23
  const i32x8 z8 = {0, 0, 0, 0, 0, 0, 0, 0};
  __builtin_amdgcn_tensor_load_to_lds(g0, g1, z4, z4, z8, 0);
#else
  __builtin_amdgcn_tensor_load_to_lds(g0, g1, z4, z4, 0);
#endif
}
#endif

// ---------------------------------------------------------------------------
// 3) bf16 GEMM: C[M][N] = A[M][Kd] * Bt[N][Kd]^T   (Bt stored N-major)
//    Block 128x256, 8 waves (2x4), wave tile 64x64 = 4x4 WMMA tiles,
//    BK=64 (2 k-steps per LDS tile), double-buffered LDS fed by TDM.
//    Per k-step per wave: 16 ds_load_b128 -> 16 v_wmma (1:1).
// ---------------------------------------------------------------------------
#define BM 128
#define BN 256
#define BK 64
#define LDSW 72  // BK + 8 pad -> 144B row stride (16B aligned) = TDM pad 32dw + 4dw

template <bool FINAL>
__global__ __launch_bounds__(256) void gemm_bf16_kernel(
    const unsigned short* __restrict__ A,   // [M][Kd] bf16 row-major
    const unsigned short* __restrict__ Bt,  // [N][Kd] bf16 row-major (transposed B)
    void* __restrict__ Cout,
    const float* __restrict__ bias,
    int N, int Kd) {
  __shared__ __align__(16) unsigned short As[2][BM][LDSW];  // 36,864 B
  __shared__ __align__(16) unsigned short Bs[2][BN][LDSW];  // 73,728 B

  const int tid      = threadIdx.x;
  const int wave     = tid >> 5;
  const int lane     = tid & 31;
  const int laneHalf = lane >> 4;
  const int laneMod  = lane & 15;

  const int bm = blockIdx.y * BM;
  const int bn = blockIdx.x * BN;
  const int wr = wave & 1;     // 2 waves x 64 rows
  const int wc = wave >> 1;    // 4 waves x 64 cols

  v8f acc[4][4];
  const v8f vzero = {0.f, 0.f, 0.f, 0.f, 0.f, 0.f, 0.f, 0.f};
#pragma unroll
  for (int i = 0; i < 4; ++i)
#pragma unroll
    for (int j = 0; j < 4; ++j) acc[i][j] = vzero;

#if USE_TDM
  const bool wv0 = (wave == 0);
  const int nIt = Kd / BK;
  if (wv0) {
    tdm_load_tile_2d((unsigned int)(uintptr_t)&As[0][0][0],
                     A + (size_t)bm * Kd, BM, BK, (unsigned long long)Kd, 4u, 3u);
    tdm_load_tile_2d((unsigned int)(uintptr_t)&Bs[0][0][0],
                     Bt + (size_t)bn * Kd, BN, BK, (unsigned long long)Kd, 4u, 3u);
  }
  int buf = 0;
  for (int it = 0; it < nIt; ++it) {
    if (wv0) wait_tensorcnt0();      // current buffer landed in LDS
    __syncthreads();                 // publish; all reads of buf^1 are done
    if (wv0 && (it + 1 < nIt)) {     // prefetch next tile into the other buffer
      const size_t k0n = (size_t)(it + 1) * BK;
      tdm_load_tile_2d((unsigned int)(uintptr_t)&As[buf ^ 1][0][0],
                       A + (size_t)bm * Kd + k0n, BM, BK, (unsigned long long)Kd, 4u, 3u);
      tdm_load_tile_2d((unsigned int)(uintptr_t)&Bs[buf ^ 1][0][0],
                       Bt + (size_t)bn * Kd + k0n, BN, BK, (unsigned long long)Kd, 4u, 3u);
    }
#else
  for (int k0 = 0; k0 < Kd; k0 += BK) {
    const int buf = 0;
    {
      const int r  = tid >> 1;
      const int cs = (tid & 1) * 32;
#pragma unroll
      for (int j = 0; j < 4; ++j)
        *(u32x4*)&As[0][r][cs + j * 8] =
            *(const u32x4*)(A + (size_t)(bm + r) * Kd + k0 + cs + j * 8);
#pragma unroll
      for (int j = 0; j < 8; ++j)
        *(u32x4*)&Bs[0][tid][j * 8] =
            *(const u32x4*)(Bt + (size_t)(bn + tid) * Kd + k0 + j * 8);
    }
    __syncthreads();
#endif

    // ---- 2 k-steps of 16 WMMAs each out of the current buffer ----
#pragma unroll
    for (int kk = 0; kk < BK; kk += 32) {
      FragBF b[4];
#pragma unroll
      for (int ni = 0; ni < 4; ++ni) {
        const int col = wc * 64 + ni * 16 + laneMod;
        b[ni].q[0] = *(const u32x4*)&Bs[buf][col][kk + laneHalf * 16];
        b[ni].q[1] = *(const u32x4*)&Bs[buf][col][kk + laneHalf * 16 + 8];
      }
#pragma unroll
      for (int mi = 0; mi < 4; ++mi) {
        FragBF a;
        const int row = wr * 64 + mi * 16 + laneMod;
        a.q[0] = *(const u32x4*)&As[buf][row][kk + laneHalf * 16];
        a.q[1] = *(const u32x4*)&As[buf][row][kk + laneHalf * 16 + 8];
#pragma unroll
        for (int ni = 0; ni < 4; ++ni)
          acc[mi][ni] = __builtin_amdgcn_wmma_f32_16x16x32_bf16(
              false, a.v, false, b[ni].v, (short)0, acc[mi][ni], false, false);
      }
    }

#if USE_TDM
    buf ^= 1;
  }
#else
    __syncthreads();
  }
#endif

  // ---- epilogue: C/D layout -> row = base + laneHalf*8 + v, col = laneMod ----
#pragma unroll
  for (int mi = 0; mi < 4; ++mi) {
#pragma unroll
    for (int ni = 0; ni < 4; ++ni) {
      const int col = bn + wc * 64 + ni * 16 + laneMod;
      if (FINAL) {
        float* C = (float*)Cout;
        const float bv = bias[col];
#pragma unroll
        for (int v = 0; v < 8; ++v) {
          const int row = bm + wr * 64 + mi * 16 + laneHalf * 8 + v;
          C[(size_t)row * N + col] = (acc[mi][ni][v] + bv) * 0.125f;  // 1/(2L), L=4
        }
      } else {
        unsigned short* C = (unsigned short*)Cout;
#pragma unroll
        for (int v = 0; v < 8; ++v) {
          const int row = bm + wr * 64 + mi * 16 + laneHalf * 8 + v;
          C[(size_t)row * N + col] = f2bf(acc[mi][ni][v]);
        }
      }
    }
  }
}

// ---------------------------------------------------------------------------
// Launch: hin->bf16; pack W1^T, W2^T; GEMM1 (C1 = hin@W1); GEMM2 (+bias, /8)
// ---------------------------------------------------------------------------
extern "C" void kernel_launch(void* const* d_in, const int* in_sizes, int n_in,
                              void* d_out, int out_size, void* d_ws, size_t ws_size,
                              hipStream_t stream) {
  const float* hin  = (const float*)d_in[0];
  const float* S1s  = (const float*)d_in[1];  // [L, Din, K]
  const float* S2s  = (const float*)d_in[2];  // [L, K, Dout]
  const float* U1s  = (const float*)d_in[3];  // [L, K, Dout]
  const float* U2s  = (const float*)d_in[4];  // [L, Din, K]
  const float* bias = (const float*)d_in[5];  // [Dout]

  const int Bm = 4096, Din = 4096, Dout = 4096, K = 256, L = 4;
  const int N1 = 2 * L * K;  // 2048 concatenated rank-K columns

  unsigned short* hinb = (unsigned short*)d_ws;              // [Bm][Din]
  unsigned short* W1T  = hinb + (size_t)Bm * Din;            // [N1][Din]
  unsigned short* W2T  = W1T  + (size_t)N1 * Din;            // [Dout][N1]
  unsigned short* C1   = W2T  + (size_t)Dout * N1;           // [Bm][N1] bf16

  cvt_bf16_kernel<<<(unsigned)(((size_t)Bm * Din) / (256 * 8)), 256, 0, stream>>>(hin, hinb);

  // W1^T[n][d]: n = l*2K + p*K + k; p=0 -> S1s[l][d][k], p=1 -> U2s[l][d][k]
  for (int l = 0; l < L; ++l)
    for (int p = 0; p < 2; ++p) {
      const float* src = (p == 0 ? S1s : U2s) + (size_t)l * Din * K;  // [Din][K]
      transpose_pack_kernel<<<dim3(Din / 32, K / 32), 256, 0, stream>>>(
          src, W1T, K, Din, l * 2 * K + p * K, 0);
    }
  // W2^T[o][n]: p=0 -> U1s[l][k][o], p=1 -> S2s[l][k][o]
  for (int l = 0; l < L; ++l)
    for (int p = 0; p < 2; ++p) {
      const float* src = (p == 0 ? U1s : S2s) + (size_t)l * K * Dout;  // [K][Dout]
      transpose_pack_kernel<<<dim3(K / 32, Dout / 32), 256, 0, stream>>>(
          src, W2T, Dout, N1, 0, l * 2 * K + p * K);
    }

  // C1[Bm][N1] = hin @ W1   (bf16 out, fp32 accumulate)
  gemm_bf16_kernel<false><<<dim3(N1 / BN, Bm / BM), 256, 0, stream>>>(
      hinb, W1T, (void*)C1, nullptr, N1, Din);

  // out[Bm][Dout] = (C1 @ W2 + bias) * 0.125
  gemm_bf16_kernel<true><<<dim3(Dout / BN, Bm / BM), 256, 0, stream>>>(
      C1, W2T, d_out, bias, Dout, N1);
}
